// MonotonicConcatAttention_72902774882614
// MI455X (gfx1250) — matrix operand order repaired and checked by reading, not compiled, not run on hardware
//
#include <hip/hip_runtime.h>
#include <hip/hip_bf16.h>
#include <math.h>

// Problem dims (match reference setup_inputs)
#define WINDOW_SZ 10
#define L_DIM 2
#define B_DIM 128
#define H_DIM 1024
#define S_DIM 1024
#define W_MAX (2 * WINDOW_SZ)      // 20
#define NEG_INF_F (-1e10f)
#define NT 16                      // N-groups of 64 (H_DIM/64)
#define KSTEPS (H_DIM / 32)        // 32 K iterations

typedef __attribute__((ext_vector_type(16))) __bf16 v16bf;
typedef __attribute__((ext_vector_type(8)))  float  v8f;

// Exact param types for __builtin_amdgcn_global_load_async_to_lds_b128
// (from hipcc diagnostic: arg0 is "int __attribute__((vector_size(16))) __device__ *")
typedef int v4i_async __attribute__((vector_size(16)));
typedef __attribute__((address_space(1))) v4i_async* gptr_v4i;
typedef __attribute__((address_space(3))) v4i_async* lptr_v4i;

#if defined(__has_builtin)
#  if __has_builtin(__builtin_amdgcn_global_load_async_to_lds_b128) && \
      __has_builtin(__builtin_amdgcn_s_wait_asynccnt)
#    define HAVE_ASYNC_LDS 1
#  endif
#endif
#ifndef HAVE_ASYNC_LDS
#  define HAVE_ASYNC_LDS 0
#endif

union Frag32 {            // 16 bf16 = 32 bytes, read as two 16B chunks
    v16bf v;
    uint4 q[2];
};

__device__ __forceinline__ void window_bounds(int t, int* lo_out, int* w_out) {
    int lo, hi;
    if (t - WINDOW_SZ > 0) {
        lo = t - WINDOW_SZ;
        hi = (t + WINDOW_SZ > S_DIM) ? S_DIM : (t + WINDOW_SZ);
    } else if (t + WINDOW_SZ < S_DIM) {
        lo = 0; hi = t + WINDOW_SZ;
    } else {
        lo = 0; hi = S_DIM;
    }
    *lo_out = lo;
    *w_out = hi - lo;
}

// ---------------------------------------------------------------------------
// Kernel 0: f32 -> bf16 conversion of W_dense [1024,2048], dh (last decoder
// slice [128,1024]) and the enc window (contiguous [w,128,1024]).
// ---------------------------------------------------------------------------
__global__ void cvt_bf16_kernel(const float* __restrict__ Wd,
                                const float* __restrict__ dec,
                                const float* __restrict__ enc,
                                const int* __restrict__ tptr,
                                __hip_bfloat16* __restrict__ Wb,
                                __hip_bfloat16* __restrict__ dhb,
                                __hip_bfloat16* __restrict__ encb) {
    int lo, w;
    window_bounds(*tptr, &lo, &w);
    const int nW   = H_DIM * 2 * H_DIM;
    const int ndh  = B_DIM * H_DIM;
    const int nenc = w * B_DIM * H_DIM;
    const int total = nW + ndh + nenc;
    const size_t enc_base = (size_t)lo * B_DIM * H_DIM;

    for (int i = blockIdx.x * blockDim.x + threadIdx.x; i < total;
         i += gridDim.x * blockDim.x) {
        if (i < nW) {
            Wb[i] = __float2bfloat16(Wd[i]);
        } else if (i < nW + ndh) {
            int j = i - nW;
            dhb[j] = __float2bfloat16(dec[(L_DIM - 1) * B_DIM * H_DIM + j]);
        } else {
            int j = i - nW - ndh;
            encb[j] = __float2bfloat16(enc[enc_base + j]);
        }
    }
}

// ---------------------------------------------------------------------------
// Kernel 1: P[b,h] = sum_d dh[b,d] * W_dense[h,d]   (K-offset 0, tiny GEMM)
// One wave per 16Mx64N tile, direct global-load WMMA.
// ---------------------------------------------------------------------------
__global__ void pgemm_kernel(const __hip_bfloat16* __restrict__ dhb,
                             const __hip_bfloat16* __restrict__ Wb,
                             float* __restrict__ P) {
    const int MT = B_DIM / 16;                 // 8 M-tiles
    int tile = blockIdx.x * (blockDim.x >> 5) + (threadIdx.x >> 5);
    if (tile >= MT * NT) return;
    int tileM = (tile / NT) * 16;
    int tileN = (tile % NT) * 64;

    int lane = threadIdx.x & 31;
    int half = lane >> 4;
    int l16  = lane & 15;

    v8f z = {0.f, 0.f, 0.f, 0.f, 0.f, 0.f, 0.f, 0.f};
    v8f acc[4] = {z, z, z, z};

    const __hip_bfloat16* arow = dhb + (size_t)(tileM + l16) * H_DIM;

    for (int kk = 0; kk < H_DIM; kk += 32) {
        Frag32 a;
        a.q[0] = *(const uint4*)(arow + kk + half * 8);
        a.q[1] = *(const uint4*)(arow + kk + half * 8 + 16);
#pragma unroll
        for (int j = 0; j < 4; ++j) {
            const __hip_bfloat16* brow =
                Wb + (size_t)(tileN + j * 16 + l16) * (2 * H_DIM);
            Frag32 b;
            b.q[0] = *(const uint4*)(brow + kk + half * 8);
            b.q[1] = *(const uint4*)(brow + kk + half * 8 + 16);
            acc[j] = __builtin_amdgcn_wmma_f32_16x16x32_bf16(
                false, a.v, false, b.v, (short)0, acc[j], false, false);
        }
    }

#pragma unroll
    for (int j = 0; j < 4; ++j) {
        int n = tileN + j * 16 + l16;
#pragma unroll
        for (int r = 0; r < 8; ++r) {
            int m = tileM + r + half * 8;
            P[m * H_DIM + n] = acc[j][r];
        }
    }
}

// ---------------------------------------------------------------------------
// B-slab staging: 64 N-rows x 32 K of bf16 (4 KB) into LDS.
// 256 x 16B chunks, 128 threads x 2. Async (ASYNCcnt) when available.
// ---------------------------------------------------------------------------
__device__ __forceinline__ void stage_B(const __hip_bfloat16* __restrict__ Wb,
                                        int tileN, int kk,
                                        __hip_bfloat16* buf, int tid) {
#pragma unroll
    for (int u = 0; u < 2; ++u) {
        int c   = tid + u * 128;
        int row = c >> 2;                 // 0..63
        int ke  = (c & 3) * 8;            // element offset in 32-K slab
        const __hip_bfloat16* gsrc =
            Wb + (size_t)(tileN + row) * (2 * H_DIM) + H_DIM + kk + ke;
        __hip_bfloat16* dst = buf + row * 32 + ke;
#if HAVE_ASYNC_LDS
        __builtin_amdgcn_global_load_async_to_lds_b128(
            (gptr_v4i)(uintptr_t)gsrc,
            (lptr_v4i)(unsigned)(uintptr_t)dst, 0, 0);
#else
        *(uint4*)dst = *(const uint4*)gsrc;
#endif
    }
}

// ---------------------------------------------------------------------------
// Kernel 2: Q-GEMM over the enc window (K-offset H in W_dense) with fused
// epilogue hidden = tanh(Q + P); partial_score = sum_n hidden * W_v[n].
//
// Block = 4 waves sharing one 64-wide N-group; B slab double-buffered in LDS
// via async loads; each wave computes a 32Mx64N tile (8 WMMAs / K-step).
// ---------------------------------------------------------------------------
__global__ void qgemm_kernel(const __hip_bfloat16* __restrict__ encb,
                             const __hip_bfloat16* __restrict__ Wb,
                             const float* __restrict__ P,
                             const float* __restrict__ Wv,
                             const int* __restrict__ tptr,
                             float* __restrict__ partials) {
    int lo, w;
    window_bounds(*tptr, &lo, &w);
    (void)lo;

    const int blockM = (blockIdx.x / NT) * 128;
    const int tileN  = (blockIdx.x % NT) * 64;
    if (blockM >= w * B_DIM) return;          // whole block: barrier-safe

    const int tid   = threadIdx.x;
    const int waveM = blockM + (tid >> 5) * 32;
    const int lane  = tid & 31;
    const int half  = lane >> 4;
    const int l16   = lane & 15;

    __shared__ __hip_bfloat16 Bbuf[2][64 * 32];   // 2 x 4KB

    v8f z = {0.f, 0.f, 0.f, 0.f, 0.f, 0.f, 0.f, 0.f};
    v8f acc0[4] = {z, z, z, z};
    v8f acc1[4] = {z, z, z, z};

    const __hip_bfloat16* arow0 = encb + (size_t)(waveM + l16) * H_DIM;
    const __hip_bfloat16* arow1 = encb + (size_t)(waveM + 16 + l16) * H_DIM;

    stage_B(Wb, tileN, 0, Bbuf[0], tid);

    for (int it = 0; it < KSTEPS; ++it) {
        const int cur = it & 1;
        const int kk  = it * 32;
        if (it + 1 < KSTEPS) {
            stage_B(Wb, tileN, kk + 32, Bbuf[cur ^ 1], tid);
#if HAVE_ASYNC_LDS
            __builtin_amdgcn_s_wait_asynccnt(2);  // stage `cur` complete (in-order)
#endif
        } else {
#if HAVE_ASYNC_LDS
            __builtin_amdgcn_s_wait_asynccnt(0);
#endif
        }
        __syncthreads();                           // all waves' stage visible

        Frag32 a0, a1;
        a0.q[0] = *(const uint4*)(arow0 + kk + half * 8);
        a0.q[1] = *(const uint4*)(arow0 + kk + half * 8 + 16);
        a1.q[0] = *(const uint4*)(arow1 + kk + half * 8);
        a1.q[1] = *(const uint4*)(arow1 + kk + half * 8 + 16);

        const __hip_bfloat16* bbase = &Bbuf[cur][0];
#pragma unroll
        for (int j = 0; j < 4; ++j) {
            const __hip_bfloat16* brow = bbase + (j * 16 + l16) * 32;
            Frag32 b;
            b.q[0] = *(const uint4*)(brow + half * 8);
            b.q[1] = *(const uint4*)(brow + half * 8 + 16);
            acc0[j] = __builtin_amdgcn_wmma_f32_16x16x32_bf16(
                false, a0.v, false, b.v, (short)0, acc0[j], false, false);
            acc1[j] = __builtin_amdgcn_wmma_f32_16x16x32_bf16(
                false, a1.v, false, b.v, (short)0, acc1[j], false, false);
        }
        __syncthreads();                           // reads done before overwrite
    }

    // Epilogue: tanh(Q+P) * Wv, reduce the 16 N-lanes of each half-wave.
    float rs0[8] = {0.f, 0.f, 0.f, 0.f, 0.f, 0.f, 0.f, 0.f};
    float rs1[8] = {0.f, 0.f, 0.f, 0.f, 0.f, 0.f, 0.f, 0.f};
#pragma unroll
    for (int j = 0; j < 4; ++j) {
        int n = tileN + j * 16 + l16;
        float wv = Wv[n];
#pragma unroll
        for (int r = 0; r < 8; ++r) {
            {
                int m  = waveM + r + half * 8;
                int bb = m & (B_DIM - 1);
                float s = tanhf(acc0[j][r] + P[bb * H_DIM + n]) * wv;
                s += __shfl_xor(s, 1);
                s += __shfl_xor(s, 2);
                s += __shfl_xor(s, 4);
                s += __shfl_xor(s, 8);
                rs0[r] += s;
            }
            {
                int m  = waveM + 16 + r + half * 8;
                int bb = m & (B_DIM - 1);
                float s = tanhf(acc1[j][r] + P[bb * H_DIM + n]) * wv;
                s += __shfl_xor(s, 1);
                s += __shfl_xor(s, 2);
                s += __shfl_xor(s, 4);
                s += __shfl_xor(s, 8);
                rs1[r] += s;
            }
        }
    }
    if (l16 == 0) {                    // lane0 -> rows +r, lane16 -> rows +8+r
        int g = tileN >> 6;
#pragma unroll
        for (int r = 0; r < 8; ++r) {
            int m0 = waveM + r + half * 8;
            int m1 = waveM + 16 + r + half * 8;
            partials[g * (W_MAX * B_DIM) + m0] = rs0[r];
            partials[g * (W_MAX * B_DIM) + m1] = rs1[r];
        }
    }
}

// ---------------------------------------------------------------------------
// Kernel 3: per-b score assembly + mask + softmax over window + context.
// ---------------------------------------------------------------------------
__global__ void softmax_ctx_kernel(const float* __restrict__ partials,
                                   const int* __restrict__ mask,
                                   const float* __restrict__ enc,
                                   const int* __restrict__ tptr,
                                   float* __restrict__ out) {
    int lo, w;
    window_bounds(*tptr, &lo, &w);
    int b = blockIdx.x;
    int tid = threadIdx.x;

    __shared__ float sc[W_MAX];
    __shared__ float att[W_MAX];

    if (tid < w) {
        float s = 0.f;
        for (int g = 0; g < NT; ++g)
            s += partials[g * (W_MAX * B_DIM) + tid * B_DIM + b];
        if (mask[(size_t)(lo + tid) * B_DIM + b] != 0) s = NEG_INF_F;
        sc[tid] = s;
    }
    __syncthreads();

    if (tid == 0) {
        float mx = sc[0];
        for (int i = 1; i < w; ++i) mx = fmaxf(mx, sc[i]);
        float sum = 0.f;
        for (int i = 0; i < w; ++i) {
            float e = expf(sc[i] - mx);
            att[i] = e;
            sum += e;
        }
        float inv = 1.f / sum;
        for (int i = 0; i < w; ++i) att[i] *= inv;
    }
    __syncthreads();

    for (int h = tid; h < H_DIM; h += blockDim.x) {
        float acc = 0.f;
        for (int i = 0; i < w; ++i)
            acc += att[i] * enc[((size_t)(lo + i) * B_DIM + b) * H_DIM + h];
        out[b * H_DIM + h] = acc;
    }
}

// ---------------------------------------------------------------------------
// Workspace layout (bytes):
//   Wb      : bf16[1024*2048]   @ 0          (4,194,304)
//   dhb     : bf16[128*1024]    @ 4,194,304  (  262,144)
//   encb    : bf16[20*128*1024] @ 4,456,448  (5,242,880)
//   P       : f32 [128*1024]    @ 9,699,328  (  524,288)
//   partials: f32 [16*2560]     @ 10,223,616 (  163,840)
// ---------------------------------------------------------------------------
extern "C" void kernel_launch(void* const* d_in, const int* in_sizes, int n_in,
                              void* d_out, int out_size, void* d_ws, size_t ws_size,
                              hipStream_t stream) {
    const float* dec  = (const float*)d_in[0];
    const float* enc  = (const float*)d_in[1];
    const int*   msk  = (const int*)d_in[2];
    const float* Wd   = (const float*)d_in[3];
    const float* Wv   = (const float*)d_in[4];
    const int*   tptr = (const int*)d_in[5];

    char* ws = (char*)d_ws;
    __hip_bfloat16* Wb   = (__hip_bfloat16*)(ws);
    __hip_bfloat16* dhb  = (__hip_bfloat16*)(ws + 4194304);
    __hip_bfloat16* encb = (__hip_bfloat16*)(ws + 4456448);
    float* P             = (float*)(ws + 9699328);
    float* partials      = (float*)(ws + 10223616);

    hipLaunchKernelGGL(cvt_bf16_kernel, dim3(1024), dim3(256), 0, stream,
                       Wd, dec, enc, tptr, Wb, dhb, encb);
    // P-GEMM: 8 M-tiles * 16 N-groups = 128 wave-tiles, 4 waves/block
    hipLaunchKernelGGL(pgemm_kernel, dim3(32), dim3(128), 0, stream,
                       dhb, Wb, P);
    // Q-GEMM: (W_MAX*128/128 = 20 M-blocks) * 16 N-groups = 320 blocks
    hipLaunchKernelGGL(qgemm_kernel, dim3(W_MAX * NT), dim3(128), 0, stream,
                       encb, Wb, P, Wv, tptr, partials);
    hipLaunchKernelGGL(softmax_ctx_kernel, dim3(B_DIM), dim3(256), 0, stream,
                       partials, msk, enc, tptr, (float*)d_out);
}
